// SpatialGNNEncoder_83760452207323
// MI455X (gfx1250) — compile-verified, matrix operand order checked by reading, not run on hardware
//
#include <hip/hip_runtime.h>
#include <hip/hip_bf16.h>
#include <math.h>

#define NNODE 50000
#define NEDGE 800000
#define DIN   2000
#define HD    256
#define DOUT  128
#define NG    16
#define NL    3
#define KIN   2048      // padded input K (2000 x + 32 PE + 16 zero)
#define KMAIN 1984      // part of K served directly from fp32 x
#define EPSB  1e-5f

typedef __bf16 bf16;
typedef __attribute__((ext_vector_type(16))) __bf16 bf16x16;
typedef __attribute__((ext_vector_type(8)))  __bf16 bf16x8;
typedef __attribute__((ext_vector_type(4)))  __bf16 bf16x4;
typedef __attribute__((ext_vector_type(8)))  float  v8f;
typedef __attribute__((ext_vector_type(4)))  float  f32x4;

// ---- fragment loaders -------------------------------------------------
// 16-bit A/B fragment layout (ISA 7.12.2): lane<16 -> K {0..7, 16..23},
// lane>=16 -> K {8..15, 24..31} relative to k0. `p` = row + k0 + sel,
// sel = (lane>=16) ? 8 : 0.
__device__ __forceinline__ bf16x16 load_frag_bf16(const bf16* p) {
    bf16x8 a0 = *(const bf16x8*)(p);
    bf16x8 a1 = *(const bf16x8*)(p + 16);
    bf16x16 r;
#pragma unroll
    for (int i = 0; i < 8; ++i) { r[i] = a0[i]; r[i + 8] = a1[i]; }
    return r;
}

__device__ __forceinline__ bf16x16 load_frag_f32(const float* p) {
    f32x4 a0 = *(const f32x4*)(p);
    f32x4 a1 = *(const f32x4*)(p + 4);
    f32x4 a2 = *(const f32x4*)(p + 16);
    f32x4 a3 = *(const f32x4*)(p + 20);
    bf16x16 r;
#pragma unroll
    for (int i = 0; i < 4; ++i) {
        r[i]      = (bf16)a0[i];
        r[i + 4]  = (bf16)a1[i];
        r[i + 8]  = (bf16)a2[i];
        r[i + 12] = (bf16)a3[i];
    }
    return r;
}

#define WMMA_BF16(A, B, C) \
    __builtin_amdgcn_wmma_f32_16x16x32_bf16(false, (A), false, (B), (short)0, (C), false, false)

// ---- small prep kernels ----------------------------------------------
// dst[c*dstStride + r] = (r < rows) ? src[r*cols + c] : 0   (transpose+pad+bf16)
__global__ void conv_transpose_kernel(const float* __restrict__ src, bf16* __restrict__ dst,
                                      int rows, int cols, int dstStride) {
    int idx = blockIdx.x * blockDim.x + threadIdx.x;
    if (idx >= cols * dstStride) return;
    int c = idx / dstStride, r = idx - c * dstStride;
    float v = (r < rows) ? src[(size_t)r * cols + c] : 0.0f;
    dst[idx] = (bf16)v;
}

// tail[m][j]: j<16 -> x[m][1984+j]; 16<=j<48 -> PE; j>=48 -> 0
__global__ void tail_kernel(const float* __restrict__ x, const float* __restrict__ pos,
                            bf16* __restrict__ tail) {
    int idx = blockIdx.x * blockDim.x + threadIdx.x;
    if (idx >= NNODE * 64) return;
    int m = idx >> 6, j = idx & 63;
    float v = 0.0f;
    if (j < 16) {
        v = x[(size_t)m * DIN + KMAIN + j];
    } else if (j < 48) {
        int jj = j - 16;            // 0..31 within PE
        int c = jj >> 4;            // coord 0/1
        int rem = jj & 15;
        int f = rem >> 1;           // freq band 0..7
        float freq = 1.0f + 9.0f * (float)f / 7.0f;   // linspace(1,10,8)
        float ang = pos[(size_t)m * 2 + c] * freq;
        v = (rem & 1) ? cosf(ang) : sinf(ang);
    }
    tail[idx] = (bf16)v;
}

__global__ void deg_kernel(const int* __restrict__ dst, float* __restrict__ deg) {
    int e = blockIdx.x * blockDim.x + threadIdx.x;
    if (e < NEDGE) atomicAdd(&deg[dst[e]], 1.0f);
}

__global__ void invdeg_kernel(const float* __restrict__ deg, float* __restrict__ invdeg) {
    int i = blockIdx.x * blockDim.x + threadIdx.x;
    if (i < NNODE) invdeg[i] = 1.0f / fmaxf(deg[i], 1.0f);
}

// ---- GEMM 1: h = relu(concat(x,pe) @ W_in + b_in) --------------------
// One wave computes a 16-row x 128-col strip (8 accumulators = 64 VGPRs,
// no spills); grid.y = 2 covers the 256 columns. x read exactly twice.
__global__ void gemm_in_kernel(const float* __restrict__ x, const bf16* __restrict__ tail,
                               const bf16* __restrict__ WinT, const float* __restrict__ bias,
                               float* __restrict__ h, bf16* __restrict__ hb, int M) {
    int wave = threadIdx.x >> 5, lane = threadIdx.x & 31;
    int tileM = blockIdx.x * 8 + wave;
    if (tileM * 16 >= M) return;
    int ntBase = blockIdx.y * 8;
    int lanePos = lane & 15, laneHalf = lane >> 4, sel = laneHalf * 8;
    int mA = tileM * 16 + lanePos; if (mA > M - 1) mA = M - 1;
    const float* arow = x + (size_t)mA * DIN;
    const bf16*  trow = tail + (size_t)mA * 64;
    const bf16*  brow = WinT + ((size_t)ntBase * 16 + lanePos) * KIN;  // nt adds nt*16*KIN
    v8f acc[8];
#pragma unroll
    for (int nt = 0; nt < 8; ++nt) acc[nt] = (v8f){};
    for (int k0 = 0; k0 < KMAIN; k0 += 32) {
        bf16x16 a = load_frag_f32(arow + k0 + sel);
#pragma unroll
        for (int nt = 0; nt < 8; ++nt) {
            bf16x16 b = load_frag_bf16(brow + (size_t)nt * 16 * KIN + k0 + sel);
            acc[nt] = WMMA_BF16(a, b, acc[nt]);
        }
    }
#pragma unroll
    for (int k0 = 0; k0 < 64; k0 += 32) {
        bf16x16 a = load_frag_bf16(trow + k0 + sel);
#pragma unroll
        for (int nt = 0; nt < 8; ++nt) {
            bf16x16 b = load_frag_bf16(brow + (size_t)nt * 16 * KIN + KMAIN + k0 + sel);
            acc[nt] = WMMA_BF16(a, b, acc[nt]);
        }
    }
#pragma unroll
    for (int nt = 0; nt < 8; ++nt) {
        int n = (ntBase + nt) * 16 + lanePos;
        float bn = bias[n];
#pragma unroll
        for (int v = 0; v < 8; ++v) {
            int m = tileM * 16 + laneHalf * 8 + v;
            if (m < M) {
                float val = fmaxf(acc[nt][v] + bn, 0.0f);
                h[(size_t)m * HD + n]  = val;
                hb[(size_t)m * HD + n] = (bf16)val;
            }
        }
    }
}

// ---- per-edge mean aggregation (bf16 gather, fp32 atomic scatter) ----
// 64 threads/edge, 4 channels/thread -> 8B vector loads, 256B/wave txns.
__global__ void edge_agg_kernel(const int* __restrict__ src, const int* __restrict__ dstv,
                                const bf16* __restrict__ hb, float* __restrict__ agg) {
    int idx = blockIdx.x * blockDim.x + threadIdx.x;   // NEDGE*64 threads exactly
    int e = idx >> 6, t = idx & 63;
    int s = src[e], d = dstv[e];
    int c = t * 4;
    bf16x4 vsrc = *(const bf16x4*)(hb + (size_t)s * HD + c);
    float* dp = agg + (size_t)d * HD + c;
    atomicAdd(dp + 0, (float)vsrc[0]);
    atomicAdd(dp + 1, (float)vsrc[1]);
    atomicAdd(dp + 2, (float)vsrc[2]);
    atomicAdd(dp + 3, (float)vsrc[3]);
}

__global__ void agg_scale_kernel(const float* __restrict__ agg, const float* __restrict__ invdeg,
                                 bf16* __restrict__ aggb) {
    size_t idx = (size_t)blockIdx.x * blockDim.x + threadIdx.x;
    if (idx >= (size_t)NNODE * HD) return;
    int m = (int)(idx >> 8);
    aggb[idx] = (bf16)(agg[idx] * invdeg[m]);
}

// ---- layer GEMM: hnew = aggb@WlT^T + hb@WrT^T + b; BN partial sums ---
__global__ void gemm_layer_kernel(const bf16* __restrict__ aggb, const bf16* __restrict__ hb,
                                  const bf16* __restrict__ WlT, const bf16* __restrict__ WrT,
                                  const float* __restrict__ bias, float* __restrict__ hnew,
                                  float* __restrict__ bn_sum, float* __restrict__ bn_sq, int M) {
    int wave = threadIdx.x >> 5, lane = threadIdx.x & 31;
    int tileM = blockIdx.x * 8 + wave;
    if (tileM * 16 >= M) return;
    int ntBase = blockIdx.y * 8;
    int lanePos = lane & 15, laneHalf = lane >> 4, sel = laneHalf * 8;
    int mA = tileM * 16 + lanePos; if (mA > M - 1) mA = M - 1;
    const bf16* arow1 = aggb + (size_t)mA * HD;
    const bf16* arow2 = hb   + (size_t)mA * HD;
    const bf16* brow1 = WlT  + ((size_t)ntBase * 16 + lanePos) * HD;
    const bf16* brow2 = WrT  + ((size_t)ntBase * 16 + lanePos) * HD;
    v8f acc[8];
#pragma unroll
    for (int nt = 0; nt < 8; ++nt) acc[nt] = (v8f){};
#pragma unroll
    for (int k0 = 0; k0 < HD; k0 += 32) {
        bf16x16 a1 = load_frag_bf16(arow1 + k0 + sel);
        bf16x16 a2 = load_frag_bf16(arow2 + k0 + sel);
#pragma unroll
        for (int nt = 0; nt < 8; ++nt) {
            bf16x16 b1 = load_frag_bf16(brow1 + (size_t)nt * 16 * HD + k0 + sel);
            acc[nt] = WMMA_BF16(a1, b1, acc[nt]);
            bf16x16 b2 = load_frag_bf16(brow2 + (size_t)nt * 16 * HD + k0 + sel);
            acc[nt] = WMMA_BF16(a2, b2, acc[nt]);
        }
    }
#pragma unroll
    for (int nt = 0; nt < 8; ++nt) {
        int n = (ntBase + nt) * 16 + lanePos;
        float bn = bias[n];
        float s = 0.0f, s2 = 0.0f;
#pragma unroll
        for (int v = 0; v < 8; ++v) {
            int m = tileM * 16 + laneHalf * 8 + v;
            if (m < M) {
                float val = acc[nt][v] + bn;
                hnew[(size_t)m * HD + n] = val;
                s += val; s2 += val * val;
            }
        }
        atomicAdd(&bn_sum[n], s);
        atomicAdd(&bn_sq[n], s2);
    }
}

// ---- BN (batch stats) + ReLU + residual ------------------------------
__global__ void bn_apply_kernel(const float* __restrict__ hnew, const float* __restrict__ bn_sum,
                                const float* __restrict__ bn_sq, const float* __restrict__ gamma,
                                const float* __restrict__ beta, float* __restrict__ h,
                                bf16* __restrict__ hb, int M) {
    size_t idx = (size_t)blockIdx.x * blockDim.x + threadIdx.x;
    if (idx >= (size_t)M * HD) return;
    int n = (int)(idx & (HD - 1));
    float invM = 1.0f / (float)M;
    float mu  = bn_sum[n] * invM;
    float var = bn_sq[n] * invM - mu * mu;
    float val = (hnew[idx] - mu) * rsqrtf(var + EPSB) * gamma[n] + beta[n];
    val = fmaxf(val, 0.0f) + h[idx];      // relu + residual (h = pre-layer value)
    h[idx]  = val;
    hb[idx] = (bf16)val;
}

// ---- output GEMM + fused mean-pool scatter ---------------------------
__global__ void gemm_out_kernel(const bf16* __restrict__ hb, const bf16* __restrict__ WoT,
                                const float* __restrict__ bias, const int* __restrict__ batch,
                                float* __restrict__ pooled, int M) {
    int wave = threadIdx.x >> 5, lane = threadIdx.x & 31;
    int tileM = blockIdx.x * 8 + wave;
    if (tileM * 16 >= M) return;
    int lanePos = lane & 15, laneHalf = lane >> 4, sel = laneHalf * 8;
    int mA = tileM * 16 + lanePos; if (mA > M - 1) mA = M - 1;
    const bf16* arow = hb  + (size_t)mA * HD;
    const bf16* brow = WoT + (size_t)lanePos * HD;
    v8f acc[8];
#pragma unroll
    for (int nt = 0; nt < 8; ++nt) acc[nt] = (v8f){};
#pragma unroll
    for (int k0 = 0; k0 < HD; k0 += 32) {
        bf16x16 a = load_frag_bf16(arow + k0 + sel);
#pragma unroll
        for (int nt = 0; nt < 8; ++nt) {
            bf16x16 b = load_frag_bf16(brow + (size_t)nt * 16 * HD + k0 + sel);
            acc[nt] = WMMA_BF16(a, b, acc[nt]);
        }
    }
#pragma unroll
    for (int nt = 0; nt < 8; ++nt) {
        int n = nt * 16 + lanePos;
        float bn = bias[n];
#pragma unroll
        for (int v = 0; v < 8; ++v) {
            int m = tileM * 16 + laneHalf * 8 + v;
            if (m < M) {
                int g = batch[m];
                atomicAdd(&pooled[(size_t)g * DOUT + n], acc[nt][v] + bn);
            }
        }
    }
}

__global__ void cnt_kernel(const int* __restrict__ batch, float* __restrict__ cnt) {
    int i = blockIdx.x * blockDim.x + threadIdx.x;
    if (i < NNODE) atomicAdd(&cnt[batch[i]], 1.0f);
}

__global__ void finalize_kernel(const float* __restrict__ pooled, const float* __restrict__ cnt,
                                float* __restrict__ out) {
    int idx = blockIdx.x * blockDim.x + threadIdx.x;
    if (idx >= NG * DOUT) return;
    int g = idx / DOUT;
    out[idx] = pooled[idx] / fmaxf(cnt[g], 1.0f);
}

// ---- host ------------------------------------------------------------
static inline size_t alignup(size_t x) { return (x + 255) & ~(size_t)255; }

extern "C" void kernel_launch(void* const* d_in, const int* in_sizes, int n_in,
                              void* d_out, int out_size, void* d_ws, size_t ws_size,
                              hipStream_t stream) {
    (void)in_sizes; (void)n_in; (void)out_size; (void)ws_size;
    const float* x      = (const float*)d_in[0];
    const float* pos    = (const float*)d_in[1];
    const int*   ei     = (const int*)d_in[2];
    const int*   batch  = (const int*)d_in[3];
    const float* W_in   = (const float*)d_in[4];
    const float* b_in   = (const float*)d_in[5];
    const float* W_l    = (const float*)d_in[6];
    const float* W_r    = (const float*)d_in[7];
    const float* b_sage = (const float*)d_in[8];
    const float* gamma  = (const float*)d_in[9];
    const float* beta   = (const float*)d_in[10];
    const float* W_out  = (const float*)d_in[11];
    const float* b_out  = (const float*)d_in[12];
    float* out = (float*)d_out;
    const int* src = ei;
    const int* dst = ei + NEDGE;

    // workspace carve-up
    char* p = (char*)d_ws; size_t off = 0;
    auto take = [&](size_t bytes) { char* r = p + off; off = alignup(off + bytes); return r; };
    bf16* WinT  = (bf16*)take((size_t)HD * KIN * 2);
    bf16* WlT   = (bf16*)take((size_t)NL * HD * HD * 2);
    bf16* WrT   = (bf16*)take((size_t)NL * HD * HD * 2);
    bf16* WoT   = (bf16*)take((size_t)DOUT * HD * 2);
    bf16* tail  = (bf16*)take((size_t)NNODE * 64 * 2);
    float* h    = (float*)take((size_t)NNODE * HD * 4);
    bf16*  hb   = (bf16*) take((size_t)NNODE * HD * 2);
    float* agg  = (float*)take((size_t)NNODE * HD * 4);
    bf16*  aggb = (bf16*) take((size_t)NNODE * HD * 2);
    float* hnew = (float*)take((size_t)NNODE * HD * 4);
    float* deg  = (float*)take((size_t)NNODE * 4);
    float* invd = (float*)take((size_t)NNODE * 4);
    float* bns  = (float*)take((size_t)HD * 4);
    float* bnq  = (float*)take((size_t)HD * 4);
    float* pooled = (float*)take((size_t)NG * DOUT * 4);
    float* cnt  = (float*)take((size_t)NG * 4);

    hipMemsetAsync(deg, 0, (size_t)NNODE * 4, stream);
    hipMemsetAsync(pooled, 0, (size_t)NG * DOUT * 4, stream);
    hipMemsetAsync(cnt, 0, (size_t)NG * 4, stream);

    // weight transpose + bf16 (W_in padded 2032 -> 2048 K)
    conv_transpose_kernel<<<(HD * KIN) / 256, 256, 0, stream>>>(W_in, WinT, DIN + 32, HD, KIN);
    for (int i = 0; i < NL; ++i) {
        conv_transpose_kernel<<<(HD * HD) / 256, 256, 0, stream>>>(W_l + (size_t)i * HD * HD,
                                                                   WlT + (size_t)i * HD * HD, HD, HD, HD);
        conv_transpose_kernel<<<(HD * HD) / 256, 256, 0, stream>>>(W_r + (size_t)i * HD * HD,
                                                                   WrT + (size_t)i * HD * HD, HD, HD, HD);
    }
    conv_transpose_kernel<<<(DOUT * HD) / 256, 256, 0, stream>>>(W_out, WoT, HD, DOUT, HD);

    tail_kernel<<<(NNODE * 64 + 255) / 256, 256, 0, stream>>>(x, pos, tail);
    deg_kernel<<<(NEDGE + 255) / 256, 256, 0, stream>>>(dst, deg);
    invdeg_kernel<<<(NNODE + 255) / 256, 256, 0, stream>>>(deg, invd);

    int tilesM = (NNODE + 15) / 16;            // 3125
    int gemmBlocks = (tilesM + 7) / 8;         // 391 blocks, 8 waves each
    dim3 gemmGrid(gemmBlocks, 2);              // grid.y: two 128-col halves
    gemm_in_kernel<<<gemmGrid, 256, 0, stream>>>(x, tail, WinT, b_in, h, hb, NNODE);

    for (int i = 0; i < NL; ++i) {
        hipMemsetAsync(agg, 0, (size_t)NNODE * HD * 4, stream);
        hipMemsetAsync(bns, 0, (size_t)HD * 4, stream);
        hipMemsetAsync(bnq, 0, (size_t)HD * 4, stream);
        edge_agg_kernel<<<(NEDGE * 64) / 256, 256, 0, stream>>>(src, dst, hb, agg);
        agg_scale_kernel<<<(NNODE * HD + 255) / 256, 256, 0, stream>>>(agg, invd, aggb);
        gemm_layer_kernel<<<gemmGrid, 256, 0, stream>>>(aggb, hb,
                                                        WlT + (size_t)i * HD * HD,
                                                        WrT + (size_t)i * HD * HD,
                                                        b_sage + (size_t)i * HD,
                                                        hnew, bns, bnq, NNODE);
        bn_apply_kernel<<<(NNODE * HD + 255) / 256, 256, 0, stream>>>(hnew, bns, bnq,
                                                                      gamma + (size_t)i * HD,
                                                                      beta + (size_t)i * HD,
                                                                      h, hb, NNODE);
    }

    gemm_out_kernel<<<gemmBlocks, 256, 0, stream>>>(hb, WoT, b_out, batch, pooled, NNODE);
    cnt_kernel<<<(NNODE + 255) / 256, 256, 0, stream>>>(batch, cnt);
    finalize_kernel<<<(NG * DOUT + 255) / 256, 256, 0, stream>>>(pooled, cnt, out);
}
